// MTLModelCGC_graph_protein_13451837571084
// MI455X (gfx1250) — matrix operand order repaired and checked by reading, not compiled, not run on hardware
//
#include <hip/hip_runtime.h>
#include <hip/hip_bf16.h>
#include <math.h>
#include <stdint.h>

// ---------------------------------------------------------------------------
// MTL CGC graph model for MI455X (gfx1250, wave32).
// Dense GEMMs -> v_wmma_f32_16x16x32_bf16 (f32 accum), A tiles staged by the
// Tensor Data Mover (async DMA, TENSORcnt) with LDS row padding, B tiles
// staged with vectorized b128 loads + manual transpose.
// Edge softmax / segment ops -> atomics (bandwidth-bound irregular part).
// ---------------------------------------------------------------------------

typedef __attribute__((ext_vector_type(16))) __bf16 v16bf;
typedef __attribute__((ext_vector_type(8)))  float  v8f;

#if defined(__gfx1250__) && __has_builtin(__builtin_amdgcn_tensor_load_to_lds) && \
    __has_builtin(__builtin_amdgcn_s_wait_tensorcnt)
#define USE_TDM 1
#else
#define USE_TDM 0
#endif

#if USE_TDM
typedef unsigned int __attribute__((ext_vector_type(4))) v4u;
typedef int          __attribute__((ext_vector_type(4))) v4i;
typedef int          __attribute__((ext_vector_type(8))) v8i_t;

// Issue a TDM load of a [tile_rows=128] x [tile_k=32] bf16 tile (row-major,
// row stride K elements) into LDS at lds_off, padding each 64B row with 16B
// so the LDS row stride is 40 halfs (80B). Rows >= rows_avail read as zero.
__device__ __forceinline__ void tdm_load_tile_A(unsigned lds_off, const __bf16* gptr,
                                                unsigned rows_avail, unsigned K) {
  unsigned long long ga = (unsigned long long)(uintptr_t)gptr;
  v4u g0;
  g0.x = 1u;                                                     // count=1 (valid)
  g0.y = lds_off;                                                // LDS byte address
  g0.z = (unsigned)ga;                                           // global addr lo
  g0.w = (unsigned)((ga >> 32) & 0x01FFFFFFull) | (2u << 30);    // addr hi | type=2
  const unsigned d0 = 32u;       // tensor_dim0 (x extent covers the 32-k tile)
  const unsigned t0 = 32u;       // tile_dim0 (k, contiguous)
  const unsigned t1 = 128u;      // tile_dim1 (rows)
  const unsigned long long s0 = K;  // tensor_dim0_stride (elements)
  v8i_t g1;
  g1[0] = (int)((1u << 16) |     // data_size = 1 -> 2 bytes
                (1u << 20) |     // pad_enable
                (3u << 22) |     // pad_interval = 3 -> every 16 DWORDs (64B row)
                (3u << 25));     // pad_amount  = 3 -> 4 DWORDs (16B)
  g1[1] = (int)((d0 & 0xFFFFu) << 16);                             // dim0[15:0]
  g1[2] = (int)((d0 >> 16) | ((rows_avail & 0xFFFFu) << 16));      // dim0 hi | dim1 lo
  g1[3] = (int)(((rows_avail >> 16) & 0xFFFFu) | (t0 << 16));      // dim1 hi | tile0
  g1[4] = (int)t1;                                                 // tile1 | tile2=0
  g1[5] = (int)(unsigned)(s0 & 0xFFFFFFFFull);                     // stride0 lo
  g1[6] = (int)(unsigned)((s0 >> 32) & 0xFFFFull);                 // stride0 hi | s1 lo
  g1[7] = 0;                                                       // stride1 hi
  v4i gz; gz[0] = gz[1] = gz[2] = gz[3] = 0;
#if __clang_major__ >= 23
  v8i_t gz8; gz8[0]=gz8[1]=gz8[2]=gz8[3]=gz8[4]=gz8[5]=gz8[6]=gz8[7]=0;
  __builtin_amdgcn_tensor_load_to_lds(g0, g1, gz, gz, gz8, 0);
#else
  __builtin_amdgcn_tensor_load_to_lds(g0, g1, gz, gz, 0);
#endif
}
#endif  // USE_TDM

// ----------------------------- small helpers -------------------------------

__device__ __forceinline__ void atomicMaxF(float* addr, float v) {
  if (v >= 0.f) atomicMax((int*)addr, __float_as_int(v));
  else          atomicMin((unsigned int*)addr, __float_as_uint(v));
}

__global__ void k_cast_bf16(const float* __restrict__ s, __bf16* __restrict__ d, long n) {
  long i = (long)blockIdx.x * blockDim.x + threadIdx.x;
  if (i < n) d[i] = (__bf16)s[i];
}

__global__ void k_fill(float* __restrict__ p, float v, long n) {
  long i = (long)blockIdx.x * blockDim.x + threadIdx.x;
  if (i < n) p[i] = v;
}

// o[i] = bias[i % C]
__global__ void k_init_bias(float* __restrict__ o, const float* __restrict__ b, long total, int C) {
  long i = (long)blockIdx.x * blockDim.x + threadIdx.x;
  if (i < total) o[i] = b[i % C];
}

// out[r] = dot(W[r,:], a)  (tiny: We @ a_e precompute)
__global__ void k_matvec(const float* __restrict__ W, const float* __restrict__ a,
                         float* __restrict__ o, int rows, int cols) {
  int r = blockIdx.x * blockDim.x + threadIdx.x;
  if (r < rows) {
    float s = 0.f;
    for (int c = 0; c < cols; ++c) s += W[(size_t)r * cols + c] * a[c];
    o[r] = s;
  }
}

// per-edge dot of edge_attr row with precomputed (We@a_e)
__global__ void k_edot(const float* __restrict__ eattr, const float* __restrict__ wa,
                       float* __restrict__ o, int E, int DE) {
  int e = blockIdx.x * blockDim.x + threadIdx.x;
  if (e < E) {
    float s = 0.f;
    const float* er = eattr + (size_t)e * DE;
    for (int d = 0; d < DE; ++d) s += er[d] * wa[d];
    o[e] = s;
  }
}

// per-row dots with two vectors (h@a_s, h@a_d), one wave per row
__global__ void k_rowdot2(const float* __restrict__ h, const float* __restrict__ a1,
                          const float* __restrict__ a2, float* __restrict__ o1,
                          float* __restrict__ o2, int n, int H) {
  int wid  = (blockIdx.x * blockDim.x + threadIdx.x) >> 5;
  int lane = threadIdx.x & 31;
  if (wid >= n) return;
  float s1 = 0.f, s2 = 0.f;
  const float* hr = h + (size_t)wid * H;
  for (int c = lane; c < H; c += 32) { float v = hr[c]; s1 += v * a1[c]; s2 += v * a2[c]; }
  for (int m = 16; m > 0; m >>= 1) { s1 += __shfl_xor(s1, m, 32); s2 += __shfl_xor(s2, m, 32); }
  if (lane == 0) { o1[wid] = s1; o2[wid] = s2; }
}

__global__ void k_edge_logit(const int* __restrict__ src, const int* __restrict__ dst,
                             const float* __restrict__ hs, const float* __restrict__ hd,
                             const float* __restrict__ edot, float* __restrict__ l,
                             float* __restrict__ mmax, int E) {
  int e = blockIdx.x * blockDim.x + threadIdx.x;
  if (e >= E) return;
  float v = hs[src[e]] + hd[dst[e]] + edot[e];
  v = v > 0.f ? v : 0.2f * v;          // leaky_relu(0.2)
  l[e] = v;
  atomicMaxF(&mmax[dst[e]], v);
}

__global__ void k_finite0(float* __restrict__ m, int n) {
  int i = blockIdx.x * blockDim.x + threadIdx.x;
  if (i < n) { float v = m[i]; if (!(v >= -3.3e38f)) m[i] = 0.f; }
}

__global__ void k_edge_exp(const float* __restrict__ l, const int* __restrict__ dst,
                           const float* __restrict__ mmax, float* __restrict__ den,
                           float* __restrict__ ex, int E) {
  int e = blockIdx.x * blockDim.x + threadIdx.x;
  if (e >= E) return;
  float v = __expf(l[e] - mmax[dst[e]]);
  ex[e] = v;
  atomicAdd(&den[dst[e]], v);
}

// one wave per edge: out[dst] += alpha * h[src]
__global__ void k_aggregate(const int* __restrict__ src, const int* __restrict__ dst,
                            const float* __restrict__ ex, const float* __restrict__ den,
                            const float* __restrict__ h, float* __restrict__ out,
                            int E, int C) {
  int wid  = (blockIdx.x * blockDim.x + threadIdx.x) >> 5;
  int lane = threadIdx.x & 31;
  if (wid >= E) return;
  int s = src[wid], d = dst[wid];
  float alpha = ex[wid] / (den[d] + 1e-16f);
  const float* hr = h + (size_t)s * C;
  float* orow = out + (size_t)d * C;
  for (int c = lane; c < C; c += 32) atomicAdd(&orow[c], alpha * hr[c]);
}

// per-channel sum / sumsq over rows
__global__ void k_bn_stats(const float* __restrict__ x, float* __restrict__ sums,
                           float* __restrict__ sumsq, int rows, int H) {
  int tid   = blockIdx.x * blockDim.x + threadIdx.x;
  int c     = tid % H;
  int r0    = tid / H;
  int rstep = (gridDim.x * blockDim.x) / H;
  float s = 0.f, s2 = 0.f;
  for (int r = r0; r < rows; r += rstep) {
    float v = x[(size_t)r * H + c];
    s += v; s2 += v * v;
  }
  atomicAdd(&sums[c], s);
  atomicAdd(&sumsq[c], s2);
}

// normalize + affine + leaky_relu(0.01); optional f32 and/or bf16 outputs
__global__ void k_bn_apply(const float* __restrict__ x, const float* __restrict__ sums,
                           const float* __restrict__ sumsq, const float* __restrict__ g,
                           const float* __restrict__ b, float invN,
                           float* __restrict__ of, __bf16* __restrict__ ob,
                           long total, int H) {
  long i = (long)blockIdx.x * blockDim.x + threadIdx.x;
  if (i >= total) return;
  int c = (int)(i % H);
  float mu  = sums[c] * invN;
  float var = sumsq[c] * invN - mu * mu;
  float v = (x[i] - mu) * rsqrtf(var + 1e-5f) * g[c] + b[c];
  v = v > 0.f ? v : 0.01f * v;
  if (of) of[i] = v;
  if (ob) ob[i] = (__bf16)v;
}

// tiny gate GEMM: [N,64] @ [64,4]
__global__ void k_gate_gemm(const float* __restrict__ x, const float* __restrict__ W,
                            float* __restrict__ o, int n) {
  int idx = blockIdx.x * blockDim.x + threadIdx.x;
  if (idx >= n * 4) return;
  int r = idx >> 2, c = idx & 3;
  const float* xr = x + (size_t)r * 64;
  float s = 0.f;
  for (int k = 0; k < 64; ++k) s += xr[k] * W[k * 4 + c];
  o[idx] = s;
}

// gate softmax + CGC combine + segment-sum pool (one wave per node)
__global__ void k_combine_pool(const float* __restrict__ glog, const float* __restrict__ r0,
                               const float* __restrict__ r1, const float* __restrict__ r2,
                               const float* __restrict__ r3, const int* __restrict__ batch,
                               float* __restrict__ pooled, int n) {
  int wid  = (blockIdx.x * blockDim.x + threadIdx.x) >> 5;
  int lane = threadIdx.x & 31;
  if (wid >= n) return;
  float g0 = glog[wid * 4 + 0], g1 = glog[wid * 4 + 1];
  float g2 = glog[wid * 4 + 2], g3 = glog[wid * 4 + 3];
  float m = fmaxf(fmaxf(g0, g1), fmaxf(g2, g3));
  g0 = __expf(g0 - m); g1 = __expf(g1 - m); g2 = __expf(g2 - m); g3 = __expf(g3 - m);
  float inv = 1.f / (g0 + g1 + g2 + g3);
  g0 *= inv; g1 *= inv; g2 *= inv; g3 *= inv;
  size_t base = (size_t)wid * 128;
  float* pb = pooled + (size_t)batch[wid] * 128;
  for (int c = lane; c < 128; c += 32) {
    float v = g0 * r0[base + c] + g1 * r1[base + c] + g2 * r2[base + c] + g3 * r3[base + c];
    atomicAdd(&pb[c], v);
  }
}

// dst[c*rows + r] = src[r*cols + c]  (f32 -> bf16)
__global__ void k_transpose_cast(const float* __restrict__ src, __bf16* __restrict__ dst,
                                 int rows, int cols) {
  long i = (long)blockIdx.x * blockDim.x + threadIdx.x;
  if (i >= (long)rows * cols) return;
  int r = (int)(i / cols), c = (int)(i % cols);
  dst[(size_t)c * rows + r] = (__bf16)src[i];
}

// row softmax with scale, bf16 output (one block per row)
__global__ void k_softmax_rows(const float* __restrict__ S, __bf16* __restrict__ P,
                               int cols, float scale) {
  __shared__ float red[256];
  int row = blockIdx.x, tid = threadIdx.x;
  const float* sr = S + (size_t)row * cols;
  float mx = -1e30f;
  for (int c = tid; c < cols; c += 256) mx = fmaxf(mx, sr[c] * scale);
  red[tid] = mx; __syncthreads();
  for (int s = 128; s > 0; s >>= 1) { if (tid < s) red[tid] = fmaxf(red[tid], red[tid + s]); __syncthreads(); }
  mx = red[0]; __syncthreads();
  float sum = 0.f;
  for (int c = tid; c < cols; c += 256) sum += __expf(sr[c] * scale - mx);
  red[tid] = sum; __syncthreads();
  for (int s = 128; s > 0; s >>= 1) { if (tid < s) red[tid] += red[tid + s]; __syncthreads(); }
  float inv = 1.f / red[0];
  for (int c = tid; c < cols; c += 256)
    P[(size_t)row * cols + c] = (__bf16)(__expf(sr[c] * scale - mx) * inv);
}

// concat(pooled, ctx) -> bf16 [rows, ca+cb]
__global__ void k_concat_cast(const float* __restrict__ a, const float* __restrict__ b,
                              __bf16* __restrict__ dst, int rows, int ca, int cb) {
  long i = (long)blockIdx.x * blockDim.x + threadIdx.x;
  int cw = ca + cb;
  if (i >= (long)rows * cw) return;
  int r = (int)(i / cw), c = (int)(i % cw);
  float v = (c < ca) ? a[(size_t)r * ca + c] : b[(size_t)r * cb + (c - ca)];
  dst[i] = (__bf16)v;
}

// final head: out[row*2 + t] = dot(hh[row], w) + b  (one wave per row)
__global__ void k_head_out(const float* __restrict__ hh, const float* __restrict__ w,
                           const float* __restrict__ b, float* __restrict__ out,
                           int rows, int H, int t) {
  int wid  = (blockIdx.x * blockDim.x + threadIdx.x) >> 5;
  int lane = threadIdx.x & 31;
  if (wid >= rows) return;
  float s = 0.f;
  const float* hr = hh + (size_t)wid * H;
  for (int c = lane; c < H; c += 32) s += hr[c] * w[c];
  for (int m = 16; m > 0; m >>= 1) s += __shfl_xor(s, m, 32);
  if (lane == 0) out[wid * 2 + t] = s + b[0];
}

// ----------------------- WMMA GEMM (bf16 x bf16 -> f32) --------------------
// C[M,N] = A[M,K](row-major bf16) * B[K,N](row-major bf16) (+bias[N]).
// Block = 256 threads = 8 waves; block tile 128x128; wave tile 16x128.
// K must be a multiple of 32 (call sites: 64,128,256,512,1152); N tiles are
// full 128 at every call site (N in {128,512}).
// A tiles: TDM async DMA -> LDS (double buffered, padded rows, OOB rows = 0).
// B tiles: vectorized b128 loads, transposed into LDS.
#define BSTRIDE 40  // halfs; 80B row stride (16B aligned), padded vs banks

__global__ __launch_bounds__(256) void k_wmma_gemm(
    const __bf16* __restrict__ A, const __bf16* __restrict__ B,
    const float* __restrict__ bias, float* __restrict__ C,
    int M, int Nn, int K) {
  __shared__ __bf16 Bs[128 * BSTRIDE];
#if USE_TDM
  __shared__ __bf16 As[2][128 * BSTRIDE];
#endif

  const int tid  = threadIdx.x;
  const int lane = tid & 31;
  const int wave = tid >> 5;
  const int n0   = blockIdx.x * 128;
  const int m0   = blockIdx.y * 128;
  const int rowBase = m0 + wave * 16;
  const int asel  = (lane & 16) ? 8 : 0;   // ISA 16-bit A layout: hi lanes take K+8
  const int kbsel = (lane & 16) ? 8 : 0;   // ISA 16-bit B layout: hi lanes take K+8
  const int nk = K >> 5;

  union CF { v8f v; float e[8]; };
  union AB16 { v16bf v; uint4 q[2]; __bf16 e[16]; };

  CF acc[8];
  #pragma unroll
  for (int j = 0; j < 8; ++j)
    #pragma unroll
    for (int r = 0; r < 8; ++r) acc[j].e[r] = 0.f;

#if USE_TDM
  const unsigned as_base = (unsigned)(uintptr_t)&As[0][0];
  const unsigned rows_avail = (unsigned)((M > m0) ? (M - m0) : 0);
  if (wave == 0)
    tdm_load_tile_A(as_base, A + (size_t)m0 * K, rows_avail, (unsigned)K);
#else
  const int arow = rowBase + (lane & 15);
  const bool arow_ok = arow < M;
#endif

  for (int it = 0; it < nk; ++it) {
    const int kk = it << 5;

    // ---- stage B[kk..kk+31][n0..n0+127] transposed: Bs[n*BSTRIDE + k] ----
    {
      const int row  = tid >> 3;        // 0..31 (k within tile)
      const int col0 = (tid & 7) << 4;  // 0,16,...,112 (n within tile)
      AB16 bb;
      if (n0 + 128 <= Nn) {
        const uint4* bp = (const uint4*)(B + (size_t)(kk + row) * Nn + n0 + col0);
        bb.q[0] = bp[0];
        bb.q[1] = bp[1];
      } else {
        #pragma unroll
        for (int j = 0; j < 16; ++j) {
          int nb = n0 + col0 + j;
          bb.e[j] = (nb < Nn) ? B[(size_t)(kk + row) * Nn + nb] : (__bf16)0.f;
        }
      }
      #pragma unroll
      for (int j = 0; j < 16; ++j) Bs[(col0 + j) * BSTRIDE + row] = bb.e[j];
    }

#if USE_TDM
    // ---- prefetch next A tile via TDM, then wait for current tile ----
    if (wave == 0) {
      if (it + 1 < nk) {
        tdm_load_tile_A(as_base + (unsigned)(((it + 1) & 1) * (128 * BSTRIDE * 2)),
                        A + (size_t)m0 * K + (kk + 32), rows_avail, (unsigned)K);
        __builtin_amdgcn_s_wait_tensorcnt(1);   // current tile complete
      } else {
        __builtin_amdgcn_s_wait_tensorcnt(0);
      }
    }
#endif
    __syncthreads();

    // ---- A fragment: lane holds M=lane%16; K chunks {asel..+7, asel+16..+23}
    AB16 af;
#if USE_TDM
    {
      const uint4* ap =
          (const uint4*)(&As[it & 1][(wave * 16 + (lane & 15)) * BSTRIDE + asel]);
      af.q[0] = ap[0];
      af.q[1] = ap[2];
    }
#else
    if (arow_ok) {
      const uint4* ap = (const uint4*)(A + (size_t)arow * K + kk + asel);
      af.q[0] = ap[0];
      af.q[1] = ap[2];
    } else {
      af.q[0] = make_uint4(0u, 0u, 0u, 0u);
      af.q[1] = make_uint4(0u, 0u, 0u, 0u);
    }
#endif

    #pragma unroll
    for (int j = 0; j < 8; ++j) {
      AB16 bfr;
      const uint4* bp = (const uint4*)(&Bs[(j * 16 + (lane & 15)) * BSTRIDE + kbsel]);
      bfr.q[0] = bp[0];
      bfr.q[1] = bp[2];
      acc[j].v = __builtin_amdgcn_wmma_f32_16x16x32_bf16(
          false, af.v, false, bfr.v, (short)0, acc[j].v, false, false);
    }
    __syncthreads();
  }

  // C/D layout: VGPR r -> M = r (lanes 0-15) / 8+r (lanes 16-31); N = lane%16
  const int mlo = (lane & 16) ? 8 : 0;
  const int cn  = lane & 15;
  #pragma unroll
  for (int j = 0; j < 8; ++j) {
    int col = n0 + j * 16 + cn;
    if (col >= Nn) continue;
    float bv = bias ? bias[col] : 0.f;
    #pragma unroll
    for (int r = 0; r < 8; ++r) {
      int row = rowBase + mlo + r;
      if (row < M) C[(size_t)row * Nn + col] = acc[j].e[r] + bv;
    }
  }
}

// ------------------------------ host driver --------------------------------

extern "C" void kernel_launch(void* const* d_in, const int* in_sizes, int n_in,
                              void* d_out, int out_size, void* d_ws, size_t ws_size,
                              hipStream_t stream) {
  const int N_ = 200000, E_ = 800000, B_ = 4096, L_ = 512;
  const int DIN_ = 64, DE_ = 16, H_ = 128, PD_ = 1152;

  const float* x     = (const float*)d_in[0];
  const int*   src   = (const int*)d_in[1];
  const int*   dst   = src + E_;
  const float* eattr = (const float*)d_in[2];
  const int*   batch = (const int*)d_in[3];
  const float* eW1   = (const float*)d_in[4];
  const float* ea_s1 = (const float*)d_in[5];
  const float* ea_d1 = (const float*)d_in[6];
  const float* eWe1  = (const float*)d_in[7];
  const float* ea_e1 = (const float*)d_in[8];
  const float* eb1   = (const float*)d_in[9];
  const float* bn1_g = (const float*)d_in[10];
  const float* bn1_b = (const float*)d_in[11];
  const float* eW2   = (const float*)d_in[12];
  const float* ea_s2 = (const float*)d_in[13];
  const float* ea_d2 = (const float*)d_in[14];
  const float* eWe2  = (const float*)d_in[15];
  const float* ea_e2 = (const float*)d_in[16];
  const float* eb2   = (const float*)d_in[17];
  const float* bn2_g = (const float*)d_in[18];
  const float* bn2_b = (const float*)d_in[19];
  const float* gW    = (const float*)d_in[20];
  const float* ga_s  = (const float*)d_in[21];
  const float* ga_d  = (const float*)d_in[22];
  const float* gWe   = (const float*)d_in[23];
  const float* ga_e  = (const float*)d_in[24];
  const float* gb    = (const float*)d_in[25];
  const float* Wq    = (const float*)d_in[26];
  const float* Wk    = (const float*)d_in[27];
  const float* Wv    = (const float*)d_in[28];
  const float* pe    = (const float*)d_in[29];
  const float* Wh1   = (const float*)d_in[30];
  const float* bh1   = (const float*)d_in[31];
  const float* hbn_g = (const float*)d_in[32];
  const float* hbn_b = (const float*)d_in[33];
  const float* Wh2   = (const float*)d_in[34];
  const float* bh2   = (const float*)d_in[35];
  float* out = (float*)d_out;

  // workspace bump allocator
  char* wsp = (char*)d_ws;
  size_t off = 0;
  auto alloc = [&](size_t bytes) -> void* {
    void* p = wsp + off;
    off = (off + bytes + 255) & ~(size_t)255;
    return p;
  };

  __bf16* xb    = (__bf16*)alloc((size_t)N_ * DIN_ * 2);
  __bf16* eW1b  = (__bf16*)alloc((size_t)6 * DIN_ * H_ * 2);
  __bf16* eW2b  = (__bf16*)alloc((size_t)6 * H_ * H_ * 2);
  __bf16* peb   = (__bf16*)alloc((size_t)2 * L_ * PD_ * 2);
  __bf16* Wqb   = (__bf16*)alloc((size_t)2 * H_ * H_ * 2);
  __bf16* Wkb   = (__bf16*)alloc((size_t)2 * PD_ * H_ * 2);
  __bf16* Wvb   = (__bf16*)alloc((size_t)2 * PD_ * H_ * 2);
  __bf16* Wh1b  = (__bf16*)alloc((size_t)2 * 2 * H_ * H_ * 2);
  float*  g1    = (float*)alloc((size_t)N_ * H_ * 4);
  float*  agg   = (float*)alloc((size_t)N_ * H_ * 4);
  __bf16* hb    = (__bf16*)alloc((size_t)N_ * H_ * 2);
  float*  reps  = (float*)alloc((size_t)6 * N_ * H_ * 4);
  float*  hs    = (float*)alloc((size_t)N_ * 4);
  float*  hd    = (float*)alloc((size_t)N_ * 4);
  float*  edot  = (float*)alloc((size_t)E_ * 4);
  float*  lbuf  = (float*)alloc((size_t)E_ * 4);
  float*  exb   = (float*)alloc((size_t)E_ * 4);
  float*  mmax  = (float*)alloc((size_t)N_ * 4);
  float*  den   = (float*)alloc((size_t)N_ * 4);
  float*  weae  = (float*)alloc((size_t)DE_ * 4);
  float*  stats = (float*)alloc((size_t)2 * H_ * 4);
  float*  hg    = (float*)alloc((size_t)N_ * 4 * 4);
  float*  glog  = (float*)alloc((size_t)N_ * 4 * 4);
  float*  pooled  = (float*)alloc((size_t)B_ * H_ * 4);
  __bf16* pooledb = (__bf16*)alloc((size_t)B_ * H_ * 2);
  float*  Km    = (float*)alloc((size_t)L_ * H_ * 4);
  float*  Vm    = (float*)alloc((size_t)L_ * H_ * 4);
  __bf16* KTb   = (__bf16*)alloc((size_t)H_ * L_ * 2);
  __bf16* Vb    = (__bf16*)alloc((size_t)L_ * H_ * 2);
  float*  Qm    = (float*)alloc((size_t)B_ * H_ * 4);
  __bf16* Qb    = (__bf16*)alloc((size_t)B_ * H_ * 2);
  float*  Sm    = (float*)alloc((size_t)B_ * L_ * 4);
  __bf16* Pb    = (__bf16*)alloc((size_t)B_ * L_ * 2);
  float*  ctx   = (float*)alloc((size_t)B_ * H_ * 4);
  __bf16* fusedb = (__bf16*)alloc((size_t)B_ * 2 * H_ * 2);
  float*  hh    = (float*)alloc((size_t)B_ * H_ * 4);

  auto cast = [&](const float* s, __bf16* d, long n) {
    k_cast_bf16<<<(unsigned)((n + 255) / 256), 256, 0, stream>>>(s, d, n);
  };
  auto fill = [&](float* p, float v, long n) {
    k_fill<<<(unsigned)((n + 255) / 256), 256, 0, stream>>>(p, v, n);
  };
  auto gemm = [&](const __bf16* A, const __bf16* Bm, const float* bias, float* C,
                  int M, int Nc, int K) {
    dim3 g((Nc + 127) / 128, (M + 127) / 128);
    k_wmma_gemm<<<g, 256, 0, stream>>>(A, Bm, bias, C, M, Nc, K);
  };
  // full GAT edge pipeline on node features h[N,C]
  auto run_gat = [&](const float* h, int C, const float* a_s, const float* a_d,
                     const float* We, const float* a_e, const float* bias, float* o) {
    k_matvec<<<1, DE_, 0, stream>>>(We, a_e, weae, DE_, C);
    k_edot<<<(E_ + 255) / 256, 256, 0, stream>>>(eattr, weae, edot, E_, DE_);
    k_rowdot2<<<((size_t)N_ * 32 + 255) / 256, 256, 0, stream>>>(h, a_s, a_d, hs, hd, N_, C);
    fill(mmax, -INFINITY, N_);
    fill(den, 0.f, N_);
    k_edge_logit<<<(E_ + 255) / 256, 256, 0, stream>>>(src, dst, hs, hd, edot, lbuf, mmax, E_);
    k_finite0<<<(N_ + 255) / 256, 256, 0, stream>>>(mmax, N_);
    k_edge_exp<<<(E_ + 255) / 256, 256, 0, stream>>>(lbuf, dst, mmax, den, exb, E_);
    k_init_bias<<<(unsigned)(((long)N_ * C + 255) / 256), 256, 0, stream>>>(o, bias, (long)N_ * C, C);
    k_aggregate<<<(unsigned)(((size_t)E_ * 32 + 255) / 256), 256, 0, stream>>>(src, dst, exb, den, h, o, E_, C);
  };

  // -------- stage 0: bf16 conversions of reused operands --------
  cast(x,   xb,   (long)N_ * DIN_);
  cast(eW1, eW1b, (long)6 * DIN_ * H_);
  cast(eW2, eW2b, (long)6 * H_ * H_);
  cast(pe,  peb,  (long)2 * L_ * PD_);
  cast(Wq,  Wqb,  (long)2 * H_ * H_);
  cast(Wk,  Wkb,  (long)2 * PD_ * H_);
  cast(Wv,  Wvb,  (long)2 * PD_ * H_);
  cast(Wh1, Wh1b, (long)2 * 2 * H_ * H_);

  // -------- stage 1: 6 expert GAT stacks --------
  for (int i = 0; i < 6; ++i) {
    // layer 1: h = x @ eW1[i]  (WMMA), then edge softmax + aggregate
    gemm(xb, eW1b + (size_t)i * DIN_ * H_, nullptr, g1, N_, H_, DIN_);
    run_gat(g1, H_, ea_s1 + i * H_, ea_d1 + i * H_, eWe1 + (size_t)i * DE_ * H_,
            ea_e1 + i * H_, eb1 + i * H_, agg);
    fill(stats, 0.f, 2 * H_);
    k_bn_stats<<<256, 256, 0, stream>>>(agg, stats, stats + H_, N_, H_);
    k_bn_apply<<<(unsigned)(((long)N_ * H_ + 255) / 256), 256, 0, stream>>>(
        agg, stats, stats + H_, bn1_g + i * H_, bn1_b + i * H_, 1.f / N_,
        nullptr, hb, (long)N_ * H_, H_);

    // layer 2: h = h1 @ eW2[i]  (WMMA), edge pipeline, BN -> reps[i]
    gemm(hb, eW2b + (size_t)i * H_ * H_, nullptr, g1, N_, H_, H_);
    run_gat(g1, H_, ea_s2 + i * H_, ea_d2 + i * H_, eWe2 + (size_t)i * DE_ * H_,
            ea_e2 + i * H_, eb2 + i * H_, agg);
    fill(stats, 0.f, 2 * H_);
    k_bn_stats<<<256, 256, 0, stream>>>(agg, stats, stats + H_, N_, H_);
    k_bn_apply<<<(unsigned)(((long)N_ * H_ + 255) / 256), 256, 0, stream>>>(
        agg, stats, stats + H_, bn2_g + i * H_, bn2_b + i * H_, 1.f / N_,
        reps + (size_t)i * N_ * H_, nullptr, (long)N_ * H_, H_);
  }

  const size_t NH = (size_t)N_ * H_;
  const float inv_sqrt_h = 0.08838834764831845f;  // 1/sqrt(128)

  // -------- stage 2: per-task gate + pool + cross-attention + head --------
  for (int t = 0; t < 2; ++t) {
    // gate GAT (C=4)
    k_gate_gemm<<<(N_ * 4 + 255) / 256, 256, 0, stream>>>(x, gW + (size_t)t * DIN_ * 4, hg, N_);
    run_gat(hg, 4, ga_s + t * 4, ga_d + t * 4, gWe + (size_t)t * DE_ * 4,
            ga_e + t * 4, gb + t * 4, glog);

    // CGC combine + global_add_pool
    fill(pooled, 0.f, (long)B_ * H_);
    k_combine_pool<<<(unsigned)(((size_t)N_ * 32 + 255) / 256), 256, 0, stream>>>(
        glog, reps, reps + NH, reps + (size_t)(2 + 2 * t) * NH,
        reps + (size_t)(3 + 2 * t) * NH, batch, pooled, N_);

    // cross-modal attention
    cast(pooled, pooledb, (long)B_ * H_);
    gemm(peb + (size_t)t * L_ * PD_, Wkb + (size_t)t * PD_ * H_, nullptr, Km, L_, H_, PD_);
    gemm(peb + (size_t)t * L_ * PD_, Wvb + (size_t)t * PD_ * H_, nullptr, Vm, L_, H_, PD_);
    k_transpose_cast<<<((long)L_ * H_ + 255) / 256, 256, 0, stream>>>(Km, KTb, L_, H_);
    cast(Vm, Vb, (long)L_ * H_);
    gemm(pooledb, Wqb + (size_t)t * H_ * H_, nullptr, Qm, B_, H_, H_);
    cast(Qm, Qb, (long)B_ * H_);
    gemm(Qb, KTb, nullptr, Sm, B_, L_, H_);          // Q @ K^T
    k_softmax_rows<<<B_, 256, 0, stream>>>(Sm, Pb, L_, inv_sqrt_h);
    gemm(Pb, Vb, nullptr, ctx, B_, H_, L_);          // attn @ V

    // head MLP with BN
    k_concat_cast<<<(unsigned)(((long)B_ * 2 * H_ + 255) / 256), 256, 0, stream>>>(
        pooled, ctx, fusedb, B_, H_, H_);
    gemm(fusedb, Wh1b + (size_t)t * 2 * H_ * H_, bh1 + t * H_, hh, B_, H_, 2 * H_);
    fill(stats, 0.f, 2 * H_);
    k_bn_stats<<<256, 256, 0, stream>>>(hh, stats, stats + H_, B_, H_);
    k_bn_apply<<<(unsigned)(((long)B_ * H_ + 255) / 256), 256, 0, stream>>>(
        hh, stats, stats + H_, hbn_g + t * H_, hbn_b + t * H_, 1.f / B_,
        hh, nullptr, (long)B_ * H_, H_);
    k_head_out<<<(unsigned)(((size_t)B_ * 32 + 255) / 256), 256, 0, stream>>>(
        hh, Wh2 + (size_t)t * H_, bh2 + t, out, B_, H_, t);
  }
}
